// EmbedMatcher_22686017257548
// MI455X (gfx1250) — compile-verified, matrix operand order checked by reading, not run on hardware
//
#include <hip/hip_runtime.h>
#include <hip/hip_bf16.h>
#include <math.h>

typedef __attribute__((ext_vector_type(16))) _Float16 v16h;
typedef __attribute__((ext_vector_type(8)))  float    v8f;

#define PAD_IDX 200000
#define ED   256      // EMBED_DIM
#define DM   512      // D_MODEL
#define DI   1024     // D_INNER
#define HID  1024     // LSTM hidden
#define BQ   2048
#define SS   32
#define MAXK 64

static __device__ __forceinline__ float waveRedSum(float v) {
#pragma unroll
  for (int off = 16; off; off >>= 1) v += __shfl_xor(v, off, 32);
  return v;
}
static __device__ __forceinline__ float sigf(float x) { return 1.f / (1.f + expf(-x)); }

// ---------------------------------------------------------------------------
// f32 -> f16 weight conversion
// ---------------------------------------------------------------------------
__global__ void cvt_f16_kernel(const float* __restrict__ src, _Float16* __restrict__ dst, int n) {
  int i = blockIdx.x * 256 + threadIdx.x;
  if (i < n) dst[i] = (_Float16)src[i];
}

// ---------------------------------------------------------------------------
// Fused neighbor encoder: one wave (32 lanes) per row.
// sim(64) -> top-10 -> stage valid (rel|ent) rows as 16x512 f16 in LDS ->
// WMMA f32_16x16x32_f16 against gcn W^T (16x64 register tiles, A fragment
// reused across 4 column tiles) -> leaky_relu/mask/mean -> gate -> tanh
// ---------------------------------------------------------------------------
__global__ __launch_bounds__(32) void neighbor_enc_kernel(
    const int* __restrict__ conn, const int* __restrict__ self_ids, int self_col,
    const float* __restrict__ emb, const _Float16* __restrict__ w16,
    const float* __restrict__ wb, const float* __restrict__ gb,
    const float* __restrict__ gate_w, const float* __restrict__ gate_wb,
    const float* __restrict__ gate_b, const float* __restrict__ gate_temp,
    float* __restrict__ outv, int outoff, int nrows)
{
  __shared__ float simL[MAXK];
  __shared__ int   relL[MAXK], entL[MAXK];
  __shared__ int   selr[16], sele[16];
  __shared__ int   s_cnt;
  __shared__ _Float16 Asub[16][DM + 8];   // padded stride to dodge bank conflicts
  __shared__ float aggL[ED];

  int n = blockIdx.x;
  if (n >= nrows) return;
  int lane = threadIdx.x;

  // self embedding: 8 floats / lane, columns lane*8..lane*8+7
  int sid = self_ids[n * 2 + self_col];
  float selfv[8];
  {
    const float* sp = emb + (size_t)sid * ED + lane * 8;
#pragma unroll
    for (int i = 0; i < 8; ++i) selfv[i] = sp[i];
  }
  float ns = 0.f;
#pragma unroll
  for (int i = 0; i < 8; ++i) ns += selfv[i] * selfv[i];
  ns = waveRedSum(ns);
  float inv_ns = 1.f / fmaxf(sqrtf(ns), 1e-12f);

  // cosine sims for 64 neighbors
  for (int k = 0; k < MAXK; ++k) {
    int rel = conn[(n * MAXK + k) * 2 + 0];
    int ent = conn[(n * MAXK + k) * 2 + 1];
    const float* ep = emb + (size_t)ent * ED + lane * 8;
    float d = 0.f, ne = 0.f;
#pragma unroll
    for (int i = 0; i < 8; ++i) { float e = ep[i]; d += selfv[i] * e; ne += e * e; }
    d = waveRedSum(d);
    ne = waveRedSum(ne);
    if (lane == 0) {
      float s = d * inv_ns / fmaxf(sqrtf(ne), 1e-12f);
      if (rel == PAD_IDX) s -= 1e9f;
      simL[k] = s; relL[k] = rel; entL[k] = ent;
    }
  }
  if (lane == 0) s_cnt = 0;
  __syncthreads();

  // iterative top-10 (butterfly argmax, first-occurrence tie-break)
  float v0 = simL[lane];
  float v1 = simL[lane + 32];
  for (int it = 0; it < 10; ++it) {
    float m; int mi;
    if (v0 >= v1) { m = v0; mi = lane; } else { m = v1; mi = lane + 32; }
#pragma unroll
    for (int off = 16; off; off >>= 1) {
      float om = __shfl_xor(m, off, 32);
      int  omi = __shfl_xor(mi, off, 32);
      if (om > m || (om == m && omi < mi)) { m = om; mi = omi; }
    }
    if (mi == lane)      v0 = -3e38f;
    if (mi == lane + 32) v1 = -3e38f;
    if (lane == 0 && m > -1e8f) {     // non-pad selected neighbors only
      int c = s_cnt;
      selr[c] = relL[mi]; sele[c] = entL[mi];
      s_cnt = c + 1;
    }
    __syncthreads();
  }
  int cnt = s_cnt;                    // uniform across wave
  float invD = 1.f / fmaxf((float)cnt, 1.f);

  // stage A: row r = concat(emb[rel_r], emb[ent_r]) as f16; lane covers 16 cols
  for (int r = 0; r < 16; ++r) {
    _Float16* dst = &Asub[r][lane * 16];
    if (r < cnt) {
      const float* src = (lane < 16) ? (emb + (size_t)selr[r] * ED + lane * 16)
                                     : (emb + (size_t)sele[r] * ED + (lane - 16) * 16);
#pragma unroll
      for (int i = 0; i < 16; ++i) dst[i] = (_Float16)src[i];
    } else {
#pragma unroll
      for (int i = 0; i < 16; ++i) dst[i] = (_Float16)0.f;
    }
  }
  __syncthreads();

  // 4 groups of 4 column tiles (64 cols each); K = 512 in 16 steps of 32.
  // A fragment loaded once per K-step and reused by 4 WMMAs.
  int m  = lane & 15;
  int ko = (lane < 16) ? 0 : 8;       // K sub-offset per ISA A/B 16-bit layout
  for (int ctg = 0; ctg < 4; ++ctg) {
    v8f acc0 = {}, acc1 = {}, acc2 = {}, acc3 = {};
    const _Float16* bpb = w16 + (size_t)(ctg * 64 + m) * DM + ko;
    for (int kb = 0; kb < DM; kb += 32) {
      v16h a;
      const _Float16* apx = &Asub[m][kb + ko];
#pragma unroll
      for (int i = 0; i < 8; ++i) { a[i] = apx[i]; a[8 + i] = apx[16 + i]; }
      v16h b0, b1, b2, b3;
      const _Float16* bp = bpb + kb;
#pragma unroll
      for (int i = 0; i < 8; ++i) { b0[i] = bp[i]; b0[8 + i] = bp[16 + i]; }
      bp += (size_t)16 * DM;
#pragma unroll
      for (int i = 0; i < 8; ++i) { b1[i] = bp[i]; b1[8 + i] = bp[16 + i]; }
      bp += (size_t)16 * DM;
#pragma unroll
      for (int i = 0; i < 8; ++i) { b2[i] = bp[i]; b2[8 + i] = bp[16 + i]; }
      bp += (size_t)16 * DM;
#pragma unroll
      for (int i = 0; i < 8; ++i) { b3[i] = bp[i]; b3[8 + i] = bp[16 + i]; }
      acc0 = __builtin_amdgcn_wmma_f32_16x16x32_f16(false, a, false, b0, (short)0, acc0, false, false);
      acc1 = __builtin_amdgcn_wmma_f32_16x16x32_f16(false, a, false, b1, (short)0, acc1, false, false);
      acc2 = __builtin_amdgcn_wmma_f32_16x16x32_f16(false, a, false, b2, (short)0, acc2, false, false);
      acc3 = __builtin_amdgcn_wmma_f32_16x16x32_f16(false, a, false, b3, (short)0, acc3, false, false);
    }
    auto epi = [&](v8f& acc, int t) {
      int col = ctg * 64 + t * 16 + m;
      float bv = wb[col] + gb[col];
      float part = 0.f;
#pragma unroll
      for (int r = 0; r < 8; ++r) {
        int mr = (lane < 16) ? r : (8 + r);   // C layout: lanes 0-15 rows 0-7, 16-31 rows 8-15
        float v = acc[r] + bv;
        v = (v > 0.f) ? v : 0.01f * v;        // leaky_relu
        if (mr < cnt) part += v;              // mask padded rows
      }
      part += __shfl_xor(part, 16, 32);
      if (lane < 16) aggL[col] = part * invD;
    };
    epi(acc0, 0); epi(acc1, 1); epi(acc2, 2); epi(acc3, 3);
  }
  __syncthreads();

  // gate + tanh epilogue
  float g = 0.f;
#pragma unroll
  for (int i = 0; i < 8; ++i) g += aggL[lane * 8 + i] * gate_w[lane * 8 + i];
  g = waveRedSum(g);
  float temp = fminf(fmaxf(gate_temp[0], 0.01f), 10.f);
  float gate = sigf((g + gate_wb[0] + gate_b[0]) / temp);
  float* op = outv + (size_t)n * DM + outoff + lane * 8;
#pragma unroll
  for (int i = 0; i < 8; ++i) op[i] = tanhf(selfv[i] + gate * aggL[lane * 8 + i]);
}

// ---------------------------------------------------------------------------
// Generic WMMA GEMM: C[M,N] = act(A[M,K(f32)] @ Bt[N,K(f16)]^T + bias)
// One wave per 16x64 strip (4 accumulators, A fragment reused 4x), 4 waves/blk.
// M mult of 16, N mult of 64, K mult of 32.
// ---------------------------------------------------------------------------
__global__ __launch_bounds__(128) void gemm16x64_kernel(
    const float* __restrict__ A, int lda, const _Float16* __restrict__ Bt, int ldb,
    const float* __restrict__ bias, float* __restrict__ C, int ldc,
    int M, int N, int K, int act)
{
  int wave = threadIdx.x >> 5, lane = threadIdx.x & 31;
  int tilesN = N >> 6;
  int tile = blockIdx.x * 4 + wave;
  if (tile >= (M >> 4) * tilesN) return;
  int tm = tile / tilesN, tn = tile % tilesN;

  int m  = lane & 15;
  int ko = (lane < 16) ? 0 : 8;
  const float*    ap  = A  + (size_t)(tm * 16 + m) * lda + ko;
  const _Float16* bpb = Bt + (size_t)(tn * 64 + m) * ldb + ko;

  v8f acc0 = {}, acc1 = {}, acc2 = {}, acc3 = {};
  for (int kb = 0; kb < K; kb += 32) {
    v16h a;
#pragma unroll
    for (int i = 0; i < 8; ++i) { a[i] = (_Float16)ap[kb + i]; a[8 + i] = (_Float16)ap[kb + 16 + i]; }
    v16h b0, b1, b2, b3;
    const _Float16* bp = bpb + kb;
#pragma unroll
    for (int i = 0; i < 8; ++i) { b0[i] = bp[i]; b0[8 + i] = bp[16 + i]; }
    bp += (size_t)16 * ldb;
#pragma unroll
    for (int i = 0; i < 8; ++i) { b1[i] = bp[i]; b1[8 + i] = bp[16 + i]; }
    bp += (size_t)16 * ldb;
#pragma unroll
    for (int i = 0; i < 8; ++i) { b2[i] = bp[i]; b2[8 + i] = bp[16 + i]; }
    bp += (size_t)16 * ldb;
#pragma unroll
    for (int i = 0; i < 8; ++i) { b3[i] = bp[i]; b3[8 + i] = bp[16 + i]; }
    acc0 = __builtin_amdgcn_wmma_f32_16x16x32_f16(false, a, false, b0, (short)0, acc0, false, false);
    acc1 = __builtin_amdgcn_wmma_f32_16x16x32_f16(false, a, false, b1, (short)0, acc1, false, false);
    acc2 = __builtin_amdgcn_wmma_f32_16x16x32_f16(false, a, false, b2, (short)0, acc2, false, false);
    acc3 = __builtin_amdgcn_wmma_f32_16x16x32_f16(false, a, false, b3, (short)0, acc3, false, false);
  }
  auto epi = [&](v8f& acc, int t) {
    int col = tn * 64 + t * 16 + m;
    float bv = bias ? bias[col] : 0.f;
#pragma unroll
    for (int r = 0; r < 8; ++r) {
      int mr = tm * 16 + ((lane < 16) ? r : (8 + r));
      float v = acc[r] + bv;
      if (act) v = fmaxf(v, 0.f);
      C[(size_t)mr * ldc + col] = v;
    }
  };
  epi(acc0, 0); epi(acc1, 1); epi(acc2, 2); epi(acc3, 3);
}

// residual + LayerNorm (unbiased std, eps=1e-3); one wave per 512-wide row
__global__ __launch_bounds__(128) void resid_ln_kernel(
    const float* __restrict__ Z, const float* __restrict__ X,
    const float* __restrict__ la, const float* __restrict__ lb,
    float* __restrict__ out, int rows)
{
  int wave = threadIdx.x >> 5, lane = threadIdx.x & 31;
  int row = blockIdx.x * 4 + wave;
  if (row >= rows) return;
  const float* zp = Z + (size_t)row * DM;
  const float* xp = X + (size_t)row * DM;
  float z[16]; float s = 0.f;
#pragma unroll
  for (int i = 0; i < 16; ++i) { int c = lane * 16 + i; z[i] = zp[c] + xp[c]; s += z[i]; }
  s = waveRedSum(s);
  float mu = s * (1.f / 512.f);
  float v = 0.f;
#pragma unroll
  for (int i = 0; i < 16; ++i) { float d = z[i] - mu; v += d * d; }
  v = waveRedSum(v);
  float inv = 1.f / (sqrtf(v / 511.f) + 1e-3f);
  float* op = out + (size_t)row * DM;
#pragma unroll
  for (int i = 0; i < 16; ++i) { int c = lane * 16 + i; op[c] = (z[i] - mu) * inv * la[c] + lb[c]; }
}

__global__ void mean_kernel(const float* __restrict__ senc, float* __restrict__ sg) {
  int d = blockIdx.x * 256 + threadIdx.x;
  if (d >= DM) return;
  float a = 0.f;
  for (int r = 0; r < SS; ++r) a += senc[(size_t)r * DM + d];
  sg[d] = a * (1.f / (float)SS);
}

// bias2[j] = bhh[j] + sum_c whh[j, 512+c] * support_g[c]
__global__ __launch_bounds__(256) void whh_bias_kernel(
    const float* __restrict__ whh, const float* __restrict__ bhh,
    const float* __restrict__ sg, float* __restrict__ out)
{
  __shared__ float sgs[DM];
  for (int i = threadIdx.x; i < DM; i += 256) sgs[i] = sg[i];
  __syncthreads();
  int j = blockIdx.x * 256 + threadIdx.x;
  float acc = bhh[j];
  const float* wr = whh + (size_t)j * HID + DM;
  for (int c = 0; c < DM; ++c) acc += wr[c] * sgs[c];
  out[j] = acc;
}

// LSTM cell elementwise; h = qenc + h_new[:, :512]
__global__ __launch_bounds__(256) void lstm_cell_kernel(
    const float* __restrict__ xg, const float* __restrict__ G,
    const float* __restrict__ bias, const float* __restrict__ qenc,
    float* __restrict__ c, float* __restrict__ h, int first)
{
  int idx = blockIdx.x * 256 + threadIdx.x;
  if (idx >= BQ * HID) return;
  int row = idx / HID, j = idx % HID;
  size_t base = (size_t)row * (4 * HID);
  float gi = xg[base + j],           gf = xg[base + HID + j];
  float gg = xg[base + 2 * HID + j], go = xg[base + 3 * HID + j];
  if (G) {
    gi += G[base + j];           gf += G[base + HID + j];
    gg += G[base + 2 * HID + j]; go += G[base + 3 * HID + j];
  }
  gi += bias[j]; gf += bias[HID + j]; gg += bias[2 * HID + j]; go += bias[3 * HID + j];
  float co = first ? 0.f : c[idx];
  float cn = sigf(gf) * co + sigf(gi) * tanhf(gg);
  float hn = sigf(go) * tanhf(cn);
  c[idx] = cn;
  if (j < DM) h[(size_t)row * DM + j] = qenc[(size_t)row * DM + j] + hn;
}

__global__ __launch_bounds__(128) void final_dot_kernel(
    const float* __restrict__ h, const float* __restrict__ sg, float* __restrict__ out)
{
  int wave = threadIdx.x >> 5, lane = threadIdx.x & 31;
  int row = blockIdx.x * 4 + wave;
  if (row >= BQ) return;
  const float* hp = h + (size_t)row * DM;
  float s = 0.f;
#pragma unroll
  for (int i = 0; i < 16; ++i) { int c = lane * 16 + i; s += hp[c] * sg[c]; }
  s = waveRedSum(s);
  if (lane == 0) out[row] = s;
}

// ---------------------------------------------------------------------------
extern "C" void kernel_launch(void* const* d_in, const int* in_sizes, int n_in,
                              void* d_out, int out_size, void* d_ws, size_t ws_size,
                              hipStream_t stream) {
  const int*   query     = (const int*)d_in[0];
  const int*   support   = (const int*)d_in[1];
  const int*   q_l1      = (const int*)d_in[2];
  const int*   q_r1      = (const int*)d_in[3];
  const int*   s_l1      = (const int*)d_in[4];
  const int*   s_r1      = (const int*)d_in[5];
  const float* emb       = (const float*)d_in[6];
  const float* gcn_w_w   = (const float*)d_in[7];
  const float* gcn_w_b   = (const float*)d_in[8];
  const float* gcn_b     = (const float*)d_in[9];
  const float* gate_w_w  = (const float*)d_in[10];
  const float* gate_w_b  = (const float*)d_in[11];
  const float* gate_b    = (const float*)d_in[12];
  const float* gate_temp = (const float*)d_in[13];
  const float* se_w1     = (const float*)d_in[14];
  const float* se_b1     = (const float*)d_in[15];
  const float* se_w2     = (const float*)d_in[16];
  const float* se_b2     = (const float*)d_in[17];
  const float* ln_a      = (const float*)d_in[18];
  const float* ln_b      = (const float*)d_in[19];
  const float* lstm_wih  = (const float*)d_in[20];
  const float* lstm_whh  = (const float*)d_in[21];
  const float* lstm_bih  = (const float*)d_in[22];
  const float* lstm_bhh  = (const float*)d_in[23];
  float* out = (float*)d_out;

  char* p = (char*)d_ws;
  auto carve = [&](size_t bytes) -> char* {
    char* r = p; p += (bytes + 255) & ~size_t(255); return r;
  };
  _Float16* gcnW16 = (_Float16*)carve((size_t)ED * DM * 2);
  _Float16* seW1h  = (_Float16*)carve((size_t)DI * DM * 2);
  _Float16* seW2h  = (_Float16*)carve((size_t)DM * DI * 2);
  _Float16* wihh   = (_Float16*)carve((size_t)4 * HID * DM * 2);
  _Float16* whhh   = (_Float16*)carve((size_t)4 * HID * HID * 2);
  float* qvec  = (float*)carve((size_t)BQ * DM * 4);
  float* svec  = (float*)carve((size_t)SS * DM * 4);
  float* qenc  = (float*)carve((size_t)BQ * DM * 4);
  float* senc  = (float*)carve((size_t)SS * DM * 4);
  float* sg    = (float*)carve((size_t)DM * 4);
  float* h1q   = (float*)carve((size_t)BQ * DI * 4);
  float* h1s   = (float*)carve((size_t)SS * DI * 4);
  float* zq    = (float*)carve((size_t)BQ * DM * 4);
  float* zs    = (float*)carve((size_t)SS * DM * 4);
  float* xg    = (float*)carve((size_t)BQ * 4 * HID * 4);
  float* Gbuf  = (float*)carve((size_t)BQ * 4 * HID * 4);
  float* hbuf  = (float*)carve((size_t)BQ * DM * 4);
  float* cbuf  = (float*)carve((size_t)BQ * HID * 4);
  float* bias2 = (float*)carve((size_t)4 * HID * 4);

  auto cvt = [&](const float* s, _Float16* d, int n) {
    cvt_f16_kernel<<<(n + 255) / 256, 256, 0, stream>>>(s, d, n);
  };
  cvt(gcn_w_w, gcnW16, ED * DM);
  cvt(se_w1, seW1h, DI * DM);
  cvt(se_w2, seW2h, DM * DI);
  cvt(lstm_wih, wihh, 4 * HID * DM);
  cvt(lstm_whh, whhh, 4 * HID * HID);

  // neighbor encoders -> qvec [2048,512], svec [32,512]
  neighbor_enc_kernel<<<BQ, 32, 0, stream>>>(q_l1, query, 0, emb, gcnW16, gcn_w_b, gcn_b,
      gate_w_w, gate_w_b, gate_b, gate_temp, qvec, 0, BQ);
  neighbor_enc_kernel<<<BQ, 32, 0, stream>>>(q_r1, query, 1, emb, gcnW16, gcn_w_b, gcn_b,
      gate_w_w, gate_w_b, gate_b, gate_temp, qvec, ED, BQ);
  neighbor_enc_kernel<<<SS, 32, 0, stream>>>(s_l1, support, 0, emb, gcnW16, gcn_w_b, gcn_b,
      gate_w_w, gate_w_b, gate_b, gate_temp, svec, 0, SS);
  neighbor_enc_kernel<<<SS, 32, 0, stream>>>(s_r1, support, 1, emb, gcnW16, gcn_w_b, gcn_b,
      gate_w_w, gate_w_b, gate_b, gate_temp, svec, ED, SS);

  auto gemm = [&](const float* A, int lda, const _Float16* Bt, int ldb, const float* bias,
                  float* C, int ldc, int M, int N, int K, int act) {
    int tiles = (M / 16) * (N / 64);
    gemm16x64_kernel<<<(tiles + 3) / 4, 128, 0, stream>>>(A, lda, Bt, ldb, bias, C, ldc, M, N, K, act);
  };

  // support encoder on queries
  gemm(qvec, DM, seW1h, DM, se_b1, h1q, DI, BQ, DI, DM, 1);
  gemm(h1q, DI, seW2h, DI, se_b2, zq, DM, BQ, DM, DI, 0);
  resid_ln_kernel<<<(BQ + 3) / 4, 128, 0, stream>>>(zq, qvec, ln_a, ln_b, qenc, BQ);
  // support encoder on supports + mean -> support_g
  gemm(svec, DM, seW1h, DM, se_b1, h1s, DI, SS, DI, DM, 1);
  gemm(h1s, DI, seW2h, DI, se_b2, zs, DM, SS, DM, DI, 0);
  resid_ln_kernel<<<(SS + 3) / 4, 128, 0, stream>>>(zs, svec, ln_a, ln_b, senc, SS);
  mean_kernel<<<2, 256, 0, stream>>>(senc, sg);

  // LSTM: xg once; r == support_g always (single support after mean)
  gemm(qenc, DM, wihh, DM, lstm_bih, xg, 4 * HID, BQ, 4 * HID, DM, 0);
  whh_bias_kernel<<<(4 * HID) / 256, 256, 0, stream>>>(lstm_whh, lstm_bhh, sg, bias2);

  lstm_cell_kernel<<<(BQ * HID) / 256, 256, 0, stream>>>(xg, nullptr, lstm_bhh, qenc, cbuf, hbuf, 1);
  for (int s = 1; s < 4; ++s) {
    gemm(hbuf, DM, whhh, HID, nullptr, Gbuf, 4 * HID, BQ, 4 * HID, DM, 0); // h @ Whh[:, :512]^T
    lstm_cell_kernel<<<(BQ * HID) / 256, 256, 0, stream>>>(xg, Gbuf, bias2, qenc, cbuf, hbuf, 0);
  }
  final_dot_kernel<<<(BQ + 3) / 4, 128, 0, stream>>>(hbuf, sg, out);
}
